// GNN_v6_10067403342425
// MI455X (gfx1250) — compile-verified, matrix operand order checked by reading, not run on hardware
//
#include <hip/hip_runtime.h>
#include <cfloat>
#include <cstdint>
#include <cstddef>

// ---------------------------------------------------------------------------
// MI455X / gfx1250 fused PointNet-GNN.
// All MLP math runs as v_wmma_f32_16x16x32_bf16 (bf16 operands, f32 accum).
// Each wave (32 lanes) owns 32 rows (two 16-row A tiles) so every B fragment
// load is reused by two WMMAs (halves the L2 B-operand stream, which is the
// roofline bottleneck once math runs on the matrix pipes). Activations live
// in a wave-private 32x256 bf16 LDS slab; the 3-layer MLP is fully fused.
// Scatter-max aggregation uses int/uint atomic max/min punning on f32.
// ---------------------------------------------------------------------------

#define N_NODES  50000
#define N_EDGES  800000
#define N_GRAPHS 64
#define ROWS_L   (N_EDGES + N_NODES)   // 850000 message rows (edges + self loops)

typedef __attribute__((ext_vector_type(16))) __bf16 v16bf;
typedef __attribute__((ext_vector_type(8)))  __bf16 v8bf;
typedef __attribute__((ext_vector_type(8)))  float  v8f;

// wave-level LDS ordering fence (cross-lane LDS dep within one wave)
#define WAVE_LDS_SYNC() asm volatile("s_wait_dscnt 0x0" ::: "memory")

__device__ __forceinline__ v16bf cat8(v8bf a, v8bf b) {
  return __builtin_shufflevector(a, b, 0,1,2,3,4,5,6,7,8,9,10,11,12,13,14,15);
}

__device__ __forceinline__ float eluf(float x) {
  return x > 0.f ? x : (__expf(x) - 1.f);
}

// float atomic-max via signed-max / unsigned-min punning (works for any sign mix,
// dest initialized to -FLT_MAX). Lowers to global_atomic_max_i32 / min_u32.
__device__ __forceinline__ void atomicMaxF(float* p, float v) {
  if (v >= 0.f) atomicMax((int*)p, __float_as_int(v));
  else          atomicMin((unsigned int*)p, __float_as_uint(v));
}

// ---------------------------------------------------------------------------
// Wave-level GEMM layer: out[(MT*16)xNOUT] = A[(MT*16)xKP] * W[KPxNOUT] + b.
// A tiles read from LDS in the ISA 16-bit A 16x32 layout (two 16B chunks per
// lane). W pre-swizzled in global memory so each B fragment is 32 contiguous
// bytes per lane; each B fragment feeds MT WMMAs (M-tiling for B reuse).
// Emit callback receives each 16x16 f32 accumulator tile (nt, mt).
// ---------------------------------------------------------------------------
template <int KP, int MT, typename Emit>
__device__ __forceinline__ void mlp_layer(const __bf16* __restrict__ wq,
                                          const float* __restrict__ bias,
                                          const __bf16* actIn, int NOUT,
                                          Emit&& emit) {
  static_assert(KP % 32 == 0, "KP must be a multiple of 32");
  const int lane = threadIdx.x & 31;
  const int lo = lane & 15;
  const int hi = lane >> 4;

  v16bf A[MT][KP / 32];
#pragma unroll
  for (int mt = 0; mt < MT; ++mt)
#pragma unroll
    for (int kt = 0; kt < KP / 32; ++kt) {
      const __bf16* p = actIn + (mt * 16 + lo) * KP + kt * 32 + hi * 8;
      A[mt][kt] = cat8(*(const v8bf*)p, *(const v8bf*)(p + 16));
    }

  const int ntiles = NOUT >> 4;
  for (int nt = 0; nt < ntiles; ++nt) {
    float bv = bias[nt * 16 + lo];
    v8f acc[MT];
#pragma unroll
    for (int mt = 0; mt < MT; ++mt)
      acc[mt] = (v8f){bv, bv, bv, bv, bv, bv, bv, bv};
#pragma unroll
    for (int kt = 0; kt < KP / 32; ++kt) {
      const __bf16* wp = wq + (size_t)(kt * ntiles + nt) * 512 + lane * 16;
      v16bf B = cat8(*(const v8bf*)wp, *(const v8bf*)(wp + 8));
#pragma unroll
      for (int mt = 0; mt < MT; ++mt)
        acc[mt] = __builtin_amdgcn_wmma_f32_16x16x32_bf16(
            false, A[mt][kt], false, B, (short)0, acc[mt], false, false);
    }
#pragma unroll
    for (int mt = 0; mt < MT; ++mt) emit(nt, mt, acc[mt]);
  }
}

// ---------------------------------------------------------------------------
// Weight convert + swizzle: fp32 [K][N] -> bf16 fragment layout.
// Fragment f = kt*(N/16)+nt, 512 bf16 per fragment, lane l holds 16 contiguous
// bf16 matching the B-operand lane layout. K padded to Kp with zeros.
// ---------------------------------------------------------------------------
__global__ __launch_bounds__(256) void k_convert(const float* __restrict__ W,
                                                 __bf16* __restrict__ out,
                                                 int K, int Kp, int N) {
  int tid = blockIdx.x * 256 + threadIdx.x;
  int total = Kp * N;
  if (tid >= total) return;
  int f = tid >> 9;
  int l = (tid >> 4) & 31;
  int e = tid & 15;
  int ntiles = N >> 4;
  int kt = f / ntiles, nt = f - kt * ntiles;
  int hi = l >> 4, lo = l & 15;
  int kk = kt * 32 + ((e >> 3) << 4) + hi * 8 + (e & 7);
  int nn = nt * 16 + lo;
  float v = (kk < K) ? W[kk * N + nn] : 0.f;
  out[tid] = (__bf16)v;
}

__global__ __launch_bounds__(256) void k_fill(float* __restrict__ p,
                                              long long n, float v) {
  long long i = (long long)blockIdx.x * 256 + threadIdx.x;
  long long stride = (long long)gridDim.x * 256;
  for (; i < n; i += stride) p[i] = v;
}

// ---------------------------------------------------------------------------
// conv1 local: per-edge MLP 6(->32 pad)->64->128->256 + scatter-max into agg.
// Block = 64 threads = 2 waves, 32 rows per wave.
// ---------------------------------------------------------------------------
__global__ __launch_bounds__(64) void k_conv1_local(
    const float* __restrict__ x, const float* __restrict__ pos,
    const int* __restrict__ ei,
    const __bf16* __restrict__ w0, const __bf16* __restrict__ w1,
    const __bf16* __restrict__ w2,
    const float* __restrict__ b0, const float* __restrict__ b1,
    const float* __restrict__ b2,
    float* __restrict__ agg) {
  __shared__ __align__(16) __bf16 act[2][32 * 256];
  __shared__ int dstn[2][32];
  const int w = threadIdx.x >> 5;
  const int lane = threadIdx.x & 31;
  const int lo = lane & 15, hi = lane >> 4;
  __bf16* a = act[w];
  long long r0 = ((long long)blockIdx.x * 2 + w) * 32;

  {  // build: each lane owns one row
    long long e = r0 + lane;
    int d = -1;
    float in6[6] = {0.f, 0.f, 0.f, 0.f, 0.f, 0.f};
    if (e < (long long)ROWS_L) {
      int s;
      if (e < (long long)N_EDGES) { s = ei[e]; d = ei[N_EDGES + e]; }
      else                        { s = d = (int)(e - N_EDGES); }
      in6[0] = x[3 * s + 0];
      in6[1] = x[3 * s + 1];
      in6[2] = x[3 * s + 2];
      in6[3] = pos[3 * s + 0] - pos[3 * d + 0];
      in6[4] = pos[3 * s + 1] - pos[3 * d + 1];
      in6[5] = pos[3 * s + 2] - pos[3 * d + 2];
    }
    dstn[w][lane] = d;
#pragma unroll
    for (int c = 0; c < 6; ++c)  a[lane * 32 + c] = (__bf16)in6[c];
#pragma unroll
    for (int c = 6; c < 32; ++c) a[lane * 32 + c] = (__bf16)0.f;
  }
  WAVE_LDS_SYNC();

  mlp_layer<32, 2>(w0, b0, a, 64, [&](int nt, int mt, v8f v) {
#pragma unroll
    for (int r = 0; r < 8; ++r)
      a[(mt * 16 + hi * 8 + r) * 64 + nt * 16 + lo] = (__bf16)eluf(v[r]);
  });
  WAVE_LDS_SYNC();
  mlp_layer<64, 2>(w1, b1, a, 128, [&](int nt, int mt, v8f v) {
#pragma unroll
    for (int r = 0; r < 8; ++r)
      a[(mt * 16 + hi * 8 + r) * 128 + nt * 16 + lo] = (__bf16)eluf(v[r]);
  });
  WAVE_LDS_SYNC();
  mlp_layer<128, 2>(w2, b2, a, 256, [&](int nt, int mt, v8f v) {
#pragma unroll
    for (int r = 0; r < 8; ++r) {
      int node = dstn[w][mt * 16 + hi * 8 + r];
      if (node >= 0)
        atomicMaxF(&agg[(size_t)node * 256 + nt * 16 + lo], v[r]);
    }
  });
}

// ---------------------------------------------------------------------------
// conv1 global: per-node MLP 256->256->256->128 with outer ELU -> x1 (bf16).
// Re-initializes agg rows to -FLT_MAX in-flight for conv2 reuse.
// ---------------------------------------------------------------------------
__global__ __launch_bounds__(64) void k_conv1_global(
    const __bf16* __restrict__ w0, const __bf16* __restrict__ w1,
    const __bf16* __restrict__ w2,
    const float* __restrict__ b0, const float* __restrict__ b1,
    const float* __restrict__ b2,
    float* __restrict__ agg, __bf16* __restrict__ x1) {
  __shared__ __align__(16) __bf16 act[2][32 * 256];
  const int w = threadIdx.x >> 5;
  const int lane = threadIdx.x & 31;
  const int lo = lane & 15, hi = lane >> 4;
  __bf16* a = act[w];
  int n0 = (blockIdx.x * 2 + w) * 32;

  for (int idx = lane; idx < 32 * 64; idx += 32) {
    int row = idx >> 6, c4 = idx & 63;
    int node = n0 + row;
    float4 v = {0.f, 0.f, 0.f, 0.f};
    if (node < N_NODES) {
      v = *((const float4*)(agg + (size_t)node * 256) + c4);
      float4 ninf = {-FLT_MAX, -FLT_MAX, -FLT_MAX, -FLT_MAX};
      *((float4*)(agg + (size_t)node * 256) + c4) = ninf;   // reuse for conv2
    }
    a[row * 256 + c4 * 4 + 0] = (__bf16)v.x;
    a[row * 256 + c4 * 4 + 1] = (__bf16)v.y;
    a[row * 256 + c4 * 4 + 2] = (__bf16)v.z;
    a[row * 256 + c4 * 4 + 3] = (__bf16)v.w;
  }
  WAVE_LDS_SYNC();

  mlp_layer<256, 2>(w0, b0, a, 256, [&](int nt, int mt, v8f v) {
#pragma unroll
    for (int r = 0; r < 8; ++r)
      a[(mt * 16 + hi * 8 + r) * 256 + nt * 16 + lo] = (__bf16)eluf(v[r]);
  });
  WAVE_LDS_SYNC();
  mlp_layer<256, 2>(w1, b1, a, 256, [&](int nt, int mt, v8f v) {
#pragma unroll
    for (int r = 0; r < 8; ++r)
      a[(mt * 16 + hi * 8 + r) * 256 + nt * 16 + lo] = (__bf16)eluf(v[r]);
  });
  WAVE_LDS_SYNC();
  mlp_layer<256, 2>(w2, b2, a, 128, [&](int nt, int mt, v8f v) {
#pragma unroll
    for (int r = 0; r < 8; ++r) {
      int node = n0 + mt * 16 + hi * 8 + r;
      if (node < N_NODES)
        x1[(size_t)node * 128 + nt * 16 + lo] = (__bf16)eluf(v[r]);
    }
  });
}

// ---------------------------------------------------------------------------
// conv2 local: per-edge MLP cat(x1_j[128], pos_j-pos_i[3]) -> 160(pad) ->
// 256 -> 256 -> 256, scatter-max into agg.
// ---------------------------------------------------------------------------
__global__ __launch_bounds__(64) void k_conv2_local(
    const __bf16* __restrict__ x1, const float* __restrict__ pos,
    const int* __restrict__ ei,
    const __bf16* __restrict__ w0, const __bf16* __restrict__ w1,
    const __bf16* __restrict__ w2,
    const float* __restrict__ b0, const float* __restrict__ b1,
    const float* __restrict__ b2,
    float* __restrict__ agg) {
  __shared__ __align__(16) __bf16 act[2][32 * 256];
  __shared__ int dstn[2][32];
  __shared__ int srcn[2][32];
  const int w = threadIdx.x >> 5;
  const int lane = threadIdx.x & 31;
  const int lo = lane & 15, hi = lane >> 4;
  __bf16* a = act[w];
  long long r0 = ((long long)blockIdx.x * 2 + w) * 32;

  {  // build: each lane owns one row (pos-diff part + tail padding)
    long long e = r0 + lane;
    int s = -1, d = -1;
    float p3[3] = {0.f, 0.f, 0.f};
    if (e < (long long)ROWS_L) {
      if (e < (long long)N_EDGES) { s = ei[e]; d = ei[N_EDGES + e]; }
      else                        { s = d = (int)(e - N_EDGES); }
      p3[0] = pos[3 * s + 0] - pos[3 * d + 0];
      p3[1] = pos[3 * s + 1] - pos[3 * d + 1];
      p3[2] = pos[3 * s + 2] - pos[3 * d + 2];
    }
    srcn[w][lane] = s;
    dstn[w][lane] = d;
    a[lane * 160 + 128] = (__bf16)p3[0];
    a[lane * 160 + 129] = (__bf16)p3[1];
    a[lane * 160 + 130] = (__bf16)p3[2];
#pragma unroll
    for (int c = 131; c < 160; ++c) a[lane * 160 + c] = (__bf16)0.f;
  }
  WAVE_LDS_SYNC();

  // gather x1[src] rows (32 rows x 128 bf16), 16B chunks
  for (int idx = lane; idx < 32 * 16; idx += 32) {
    int row = idx >> 4, c = idx & 15;
    int s = srcn[w][row];
    v8bf val;
#pragma unroll
    for (int i = 0; i < 8; ++i) val[i] = (__bf16)0.f;
    if (s >= 0) val = *(const v8bf*)(x1 + (size_t)s * 128 + c * 8);
    *(v8bf*)(a + row * 160 + c * 8) = val;
  }
  WAVE_LDS_SYNC();

  mlp_layer<160, 2>(w0, b0, a, 256, [&](int nt, int mt, v8f v) {
#pragma unroll
    for (int r = 0; r < 8; ++r)
      a[(mt * 16 + hi * 8 + r) * 256 + nt * 16 + lo] = (__bf16)eluf(v[r]);
  });
  WAVE_LDS_SYNC();
  mlp_layer<256, 2>(w1, b1, a, 256, [&](int nt, int mt, v8f v) {
#pragma unroll
    for (int r = 0; r < 8; ++r)
      a[(mt * 16 + hi * 8 + r) * 256 + nt * 16 + lo] = (__bf16)eluf(v[r]);
  });
  WAVE_LDS_SYNC();
  mlp_layer<256, 2>(w2, b2, a, 256, [&](int nt, int mt, v8f v) {
#pragma unroll
    for (int r = 0; r < 8; ++r) {
      int node = dstn[w][mt * 16 + hi * 8 + r];
      if (node >= 0)
        atomicMaxF(&agg[(size_t)node * 256 + nt * 16 + lo], v[r]);
    }
  });
}

// ---------------------------------------------------------------------------
// conv2 global: per-node MLP 256->256->256->128, outer ELU, fused graph
// pooling (segment max / sum / count) via atomics.
// ---------------------------------------------------------------------------
__global__ __launch_bounds__(64) void k_conv2_global(
    const __bf16* __restrict__ w0, const __bf16* __restrict__ w1,
    const __bf16* __restrict__ w2,
    const float* __restrict__ b0, const float* __restrict__ b1,
    const float* __restrict__ b2,
    const float* __restrict__ agg, const int* __restrict__ batch,
    float* __restrict__ xmax, float* __restrict__ xadd,
    float* __restrict__ counts) {
  __shared__ __align__(16) __bf16 act[2][32 * 256];
  __shared__ int bat[2][32];
  const int w = threadIdx.x >> 5;
  const int lane = threadIdx.x & 31;
  const int lo = lane & 15, hi = lane >> 4;
  __bf16* a = act[w];
  int n0 = (blockIdx.x * 2 + w) * 32;

  {
    int node = n0 + lane;
    int g = -1;
    if (node < N_NODES) {
      g = batch[node];
      atomicAdd(&counts[g], 1.0f);
    }
    bat[w][lane] = g;
  }
  for (int idx = lane; idx < 32 * 64; idx += 32) {
    int row = idx >> 6, c4 = idx & 63;
    int node = n0 + row;
    float4 v = {0.f, 0.f, 0.f, 0.f};
    if (node < N_NODES) v = *((const float4*)(agg + (size_t)node * 256) + c4);
    a[row * 256 + c4 * 4 + 0] = (__bf16)v.x;
    a[row * 256 + c4 * 4 + 1] = (__bf16)v.y;
    a[row * 256 + c4 * 4 + 2] = (__bf16)v.z;
    a[row * 256 + c4 * 4 + 3] = (__bf16)v.w;
  }
  WAVE_LDS_SYNC();

  mlp_layer<256, 2>(w0, b0, a, 256, [&](int nt, int mt, v8f v) {
#pragma unroll
    for (int r = 0; r < 8; ++r)
      a[(mt * 16 + hi * 8 + r) * 256 + nt * 16 + lo] = (__bf16)eluf(v[r]);
  });
  WAVE_LDS_SYNC();
  mlp_layer<256, 2>(w1, b1, a, 256, [&](int nt, int mt, v8f v) {
#pragma unroll
    for (int r = 0; r < 8; ++r)
      a[(mt * 16 + hi * 8 + r) * 256 + nt * 16 + lo] = (__bf16)eluf(v[r]);
  });
  WAVE_LDS_SYNC();
  mlp_layer<256, 2>(w2, b2, a, 128, [&](int nt, int mt, v8f v) {
#pragma unroll
    for (int r = 0; r < 8; ++r) {
      int g = bat[w][mt * 16 + hi * 8 + r];
      if (g >= 0) {
        float val = eluf(v[r]);
        int col = nt * 16 + lo;
        atomicMaxF(&xmax[g * 128 + col], val);
        atomicAdd(&xadd[g * 128 + col], val);
      }
    }
  });
}

// ---------------------------------------------------------------------------
// Final: pooled[g] = [max(128) | mean(128) | sum(128)] @ W[384x2] + b.
// ---------------------------------------------------------------------------
__global__ __launch_bounds__(128) void k_final(
    const float* __restrict__ xmax, const float* __restrict__ xadd,
    const float* __restrict__ counts, const float* __restrict__ W,
    const float* __restrict__ b, float* __restrict__ out) {
  int t = threadIdx.x;
  if (t >= N_GRAPHS * 2) return;
  int g = t >> 1, c = t & 1;
  float cnt = fmaxf(counts[g], 1.0f);
  float acc = b[c];
  for (int k = 0; k < 128; ++k) {
    float mx = xmax[g * 128 + k];
    float sm = xadd[g * 128 + k];
    acc += mx * W[k * 2 + c] + (sm / cnt) * W[(128 + k) * 2 + c] +
           sm * W[(256 + k) * 2 + c];
  }
  out[g * 2 + c] = acc;
}

// ---------------------------------------------------------------------------
// Host launch
// ---------------------------------------------------------------------------
// workspace layout (bytes)
static const size_t OFF_WQ   = 0;          // 542720 bf16 elems (swizzled weights)
static const size_t OFF_AGG  = 1085440;    // 50000*256 f32
static const size_t OFF_X1   = 52285440;   // 50000*128 bf16
static const size_t OFF_XMAX = 65085440;   // 64*128 f32
static const size_t OFF_XADD = 65118208;   // 64*128 f32 (+counts right after)
static const size_t OFF_CNT  = 65150976;   // 64 f32

// weight sub-offsets in bf16 elems
static const size_t O_L1A = 0,      O_L1B = 2048,   O_L1C = 10240;
static const size_t O_G1A = 43008,  O_G1B = 108544, O_G1C = 174080;
static const size_t O_L2A = 206848, O_L2B = 247808, O_L2C = 313344;
static const size_t O_G2A = 378880, O_G2B = 444416, O_G2C = 509952;

extern "C" void kernel_launch(void* const* d_in, const int* in_sizes, int n_in,
                              void* d_out, int out_size, void* d_ws,
                              size_t ws_size, hipStream_t stream) {
  (void)in_sizes; (void)n_in; (void)out_size; (void)ws_size;

  const float* x   = (const float*)d_in[0];
  const float* pos = (const float*)d_in[1];
  const int*   ei  = (const int*)d_in[2];
  const int*   bat = (const int*)d_in[3];

  char* ws = (char*)d_ws;
  __bf16* wq    = (__bf16*)(ws + OFF_WQ);
  float*  agg   = (float*)(ws + OFF_AGG);
  __bf16* x1    = (__bf16*)(ws + OFF_X1);
  float*  xmax  = (float*)(ws + OFF_XMAX);
  float*  xadd  = (float*)(ws + OFF_XADD);
  float*  cnts  = (float*)(ws + OFF_CNT);

  // 1) convert + swizzle all 12 weight matrices fp32 -> bf16 fragments
  struct WT { int idx; int K; int Kp; int N; size_t off; };
  const WT tab[12] = {
      {4, 6, 32, 64, O_L1A},   {6, 64, 64, 128, O_L1B},  {8, 128, 128, 256, O_L1C},
      {10, 256, 256, 256, O_G1A}, {12, 256, 256, 256, O_G1B}, {14, 256, 256, 128, O_G1C},
      {16, 131, 160, 256, O_L2A}, {18, 256, 256, 256, O_L2B}, {20, 256, 256, 256, O_L2C},
      {22, 256, 256, 256, O_G2A}, {24, 256, 256, 256, O_G2B}, {26, 256, 256, 128, O_G2C},
  };
  for (int i = 0; i < 12; ++i) {
    int total = tab[i].Kp * tab[i].N;
    k_convert<<<(total + 255) / 256, 256, 0, stream>>>(
        (const float*)d_in[tab[i].idx], wq + tab[i].off, tab[i].K, tab[i].Kp,
        tab[i].N);
  }

  // 2) init agg (-inf) and pooling buffers (max=-inf, sum/counts=0)
  k_fill<<<2048, 256, 0, stream>>>(agg, (long long)N_NODES * 256, -FLT_MAX);
  k_fill<<<32, 256, 0, stream>>>(xmax, (long long)N_GRAPHS * 128, -FLT_MAX);
  k_fill<<<33, 256, 0, stream>>>(xadd, (long long)(N_GRAPHS * 128 + N_GRAPHS),
                                 0.0f);  // xadd + counts contiguous

  const int blocksE = (ROWS_L + 63) / 64;   // 64 rows per block (2 waves x 32)
  const int blocksN = (N_NODES + 63) / 64;

  // 3) conv1 local + scatter-max
  k_conv1_local<<<blocksE, 64, 0, stream>>>(
      x, pos, ei, wq + O_L1A, wq + O_L1B, wq + O_L1C,
      (const float*)d_in[5], (const float*)d_in[7], (const float*)d_in[9], agg);

  // 4) conv1 global -> x1 (bf16), re-inits agg rows for conv2
  k_conv1_global<<<blocksN, 64, 0, stream>>>(
      wq + O_G1A, wq + O_G1B, wq + O_G1C, (const float*)d_in[11],
      (const float*)d_in[13], (const float*)d_in[15], agg, x1);

  // 5) conv2 local + scatter-max
  k_conv2_local<<<blocksE, 64, 0, stream>>>(
      x1, pos, ei, wq + O_L2A, wq + O_L2B, wq + O_L2C,
      (const float*)d_in[17], (const float*)d_in[19], (const float*)d_in[21],
      agg);

  // 6) conv2 global + fused graph pooling
  k_conv2_global<<<blocksN, 64, 0, stream>>>(
      wq + O_G2A, wq + O_G2B, wq + O_G2C, (const float*)d_in[23],
      (const float*)d_in[25], (const float*)d_in[27], agg, bat, xmax, xadd,
      cnts);

  // 7) final linear on pooled features
  k_final<<<1, 128, 0, stream>>>(xmax, xadd, cnts, (const float*)d_in[28],
                                 (const float*)d_in[29], (float*)d_out);
}
